// MHFlashAttn_8031588843658
// MI455X (gfx1250) — compile-verified, matrix operand order checked by reading, not run on hardware
//
#include <hip/hip_runtime.h>

// Fused multi-head causal attention for gfx1250 (MI455X).
// One 256-thread block (8 wave32) per (batch, head) pair: 2048 blocks.
// All intermediates (Xh, W^T, Q, K, V^T, P) are f16 in LDS; all matrix math
// uses v_wmma_f32_16x16x32_f16 with f32 accumulation.

typedef __attribute__((ext_vector_type(16))) _Float16 v16h;
typedef __attribute__((ext_vector_type(8)))  float    v8f;

#define TDIM   128   // sequence length
#define CDIM   512   // embedding dim
#define HS     128   // head size (== C / HEADS)
#define NHEAD  4
#define PITCH  136   // LDS row pitch in halfs (128 + 8 pad: bank-conflict avoidance)

union FragU { v16h h; unsigned int u[8]; };

// Load a 16x32 f16 WMMA fragment from a row-major LDS tile.
// A-matrix layout (ISA 7.12.2): lane l -> row (row0 + (l&15)), group g=l>>4,
// VGPR v holds halfs at k = 2*(v&3) + 16*(v>>2) + 8*g (+0,+1).
// Bytes 0..15 and 32..47 of the row slice -> compiler emits 2x ds_load_b128.
// B-fragments use the transpose-symmetric layout (lane = N), so the same
// loader works when the staged tile is the transpose of the B matrix.
__device__ __forceinline__ v16h lds_frag(const _Float16* __restrict__ base,
                                         int row0, int k0) {
  const int lane = threadIdx.x & 31;
  const int m = lane & 15;
  const int g = lane >> 4;
  const _Float16* rp = base + (row0 + m) * PITCH + k0 + g * 8;
  FragU f;
#pragma unroll
  for (int v = 0; v < 8; ++v) {
    const int kb = ((v & 3) << 1) + ((v >> 2) << 4);
    f.u[v] = *(const unsigned int*)(rp + kb);   // 4B-aligned: pitch & kb even
  }
  return f.h;
}

// One 16x16 output tile of a (16 x 128) * (128 x 16) product.
// All 4 B-fragments are loaded up front so the 4 chained WMMAs overlap the
// LDS latency instead of waiting load-by-load.
__device__ __forceinline__ v8f mm_tile(const v16h a4[4],
                                       const _Float16* __restrict__ bbase,
                                       int row0) {
  v16h bf[4];
#pragma unroll
  for (int kk = 0; kk < 4; ++kk) bf[kk] = lds_frag(bbase, row0, kk * 32);
  v8f acc = {};
#pragma unroll
  for (int kk = 0; kk < 4; ++kk)
    acc = __builtin_amdgcn_wmma_f32_16x16x32_f16(
              false, a4[kk], false, bf[kk], (short)0, acc, false, false);
  return acc;
}

__global__ __launch_bounds__(256)
void MHFlashAttn_kernel(const float* __restrict__ x,
                        const float* __restrict__ Wq,
                        const float* __restrict__ Wk,
                        const float* __restrict__ Wv,
                        float* __restrict__ out) {
  __shared__ _Float16 sX [TDIM * PITCH];      // Xh tile, row-major (t, c)
  __shared__ _Float16 sW [3 * HS * PITCH];    // W^T per matrix: (d, c)
  __shared__ _Float16 sQ [TDIM * PITCH];      // Q, row-major (t, d)
  __shared__ _Float16 sK [TDIM * PITCH];      // K, row-major (t, d)
  __shared__ _Float16 sVt[HS   * PITCH];      // V^T: (d, s)
  __shared__ _Float16 sP [TDIM * PITCH];      // softmax probs, per-wave slabs

  const int bid  = blockIdx.x;           // bid = b*NHEAD + h
  const int b    = bid >> 2;
  const int h    = bid & 3;
  const int tid  = threadIdx.x;
  const int lane = tid & 31;
  const int wave = tid >> 5;             // 8 waves
  const int m    = lane & 15;
  const int g    = lane >> 4;
  const int r0   = wave * 16;            // this wave's 16-row stripe

  // ---------------- Stage Xh tile (f32 -> f16), coalesced float4 loads ----
  const float* xb = x + (size_t)b * TDIM * CDIM + h * HS;
#pragma unroll
  for (int it = 0; it < 16; ++it) {
    const int idx4 = tid * 4 + it * 1024;
    const int t = idx4 >> 7;
    const int c = idx4 & 127;
    const float4 v = *(const float4*)(xb + (size_t)t * CDIM + c);
    sX[t * PITCH + c + 0] = (_Float16)v.x;
    sX[t * PITCH + c + 1] = (_Float16)v.y;
    sX[t * PITCH + c + 2] = (_Float16)v.z;
    sX[t * PITCH + c + 3] = (_Float16)v.w;
  }
  // ---------------- Stage W^T (float4 reads over d, transposed LDS write) -
#pragma unroll 1
  for (int mat = 0; mat < 3; ++mat) {
    const float* W = (mat == 0) ? Wq : (mat == 1) ? Wk : Wv;   // (c, d) layout
    _Float16* dst = sW + mat * HS * PITCH;
#pragma unroll
    for (int it = 0; it < 16; ++it) {
      const int idx4 = tid * 4 + it * 1024;       // 128*128 elems / (256*4)
      const int c = idx4 >> 7;
      const int d = idx4 & 127;
      const float4 v = *(const float4*)(W + idx4);
      dst[(d + 0) * PITCH + c] = (_Float16)v.x;
      dst[(d + 1) * PITCH + c] = (_Float16)v.y;
      dst[(d + 2) * PITCH + c] = (_Float16)v.z;
      dst[(d + 3) * PITCH + c] = (_Float16)v.w;
    }
  }
  __syncthreads();

  // ---------------- Phase 1: Q/K/V = Xh * W  (96 WMMAs per wave) ----------
  v16h xa[4];
#pragma unroll
  for (int kk = 0; kk < 4; ++kk) xa[kk] = lds_frag(sX, r0, kk * 32);

  // Q tiles
#pragma unroll 2
  for (int n = 0; n < 8; ++n) {
    const v8f acc = mm_tile(xa, sW + 0 * HS * PITCH, n * 16);
#pragma unroll
    for (int j = 0; j < 8; ++j)     // C-layout: (row r0+j+8g, col n*16+m)
      sQ[(r0 + j + 8 * g) * PITCH + n * 16 + m] = (_Float16)acc[j];
  }
  // K tiles
#pragma unroll 2
  for (int n = 0; n < 8; ++n) {
    const v8f acc = mm_tile(xa, sW + 1 * HS * PITCH, n * 16);
#pragma unroll
    for (int j = 0; j < 8; ++j)
      sK[(r0 + j + 8 * g) * PITCH + n * 16 + m] = (_Float16)acc[j];
  }
  // V tiles, stored transposed so PV GEMM B-fragments are contiguous rows
#pragma unroll 2
  for (int n = 0; n < 8; ++n) {
    const v8f acc = mm_tile(xa, sW + 2 * HS * PITCH, n * 16);
#pragma unroll
    for (int j = 0; j < 8; ++j)     // j runs along contiguous halfs -> b128
      sVt[(n * 16 + m) * PITCH + (r0 + j + 8 * g)] = (_Float16)acc[j];
  }
  __syncthreads();

  // ---------------- Phase 2a: S = Q * K^T  (32 WMMAs per wave) ------------
  v16h qf[4];
#pragma unroll
  for (int kk = 0; kk < 4; ++kk) qf[kk] = lds_frag(sQ, r0, kk * 32);

  v8f s[8];
#pragma unroll 2
  for (int n = 0; n < 8; ++n) s[n] = mm_tile(qf, sK, n * 16);

  // ---------------- Phase 2b: causal mask + softmax (in-register) ---------
  const float scale = 0.08838834764831845f;   // 1/sqrt(128)
#pragma unroll
  for (int j = 0; j < 8; ++j) {
    const int r = r0 + j + 8 * g;             // global query row
    float mx = -3.0e38f;
#pragma unroll
    for (int n = 0; n < 8; ++n) {
      float vv = s[n][j] * scale;
      vv = ((n * 16 + m) <= r) ? vv : -3.0e38f;   // causal mask
      s[n][j] = vv;
      mx = fmaxf(mx, vv);
    }
#pragma unroll
    for (int off = 1; off < 16; off <<= 1)        // 16-lane row reduction
      mx = fmaxf(mx, __shfl_xor(mx, off, 32));
    float sum = 0.0f;
#pragma unroll
    for (int n = 0; n < 8; ++n) {
      const float e = __expf(s[n][j] - mx);
      s[n][j] = e;
      sum += e;
    }
#pragma unroll
    for (int off = 1; off < 16; off <<= 1)
      sum += __shfl_xor(sum, off, 32);
    const float inv = 1.0f / sum;
#pragma unroll
    for (int n = 0; n < 8; ++n) s[n][j] *= inv;
  }

  // Spill P (f16) into this wave's private LDS slab; same wave re-reads it,
  // so no block barrier is needed (DScnt ordering handled by the compiler).
  _Float16* pw = sP + r0 * PITCH;
#pragma unroll
  for (int n = 0; n < 8; ++n)
#pragma unroll
    for (int j = 0; j < 8; ++j)
      pw[(j + 8 * g) * PITCH + n * 16 + m] = (_Float16)s[n][j];

  // ---------------- Phase 2c: O = P * V  (32 WMMAs per wave) --------------
  v16h pf[4];
#pragma unroll
  for (int kk = 0; kk < 4; ++kk) pf[kk] = lds_frag(pw, 0, kk * 32);

  float* ob = out + (size_t)bid * TDIM * HS;   // (B,H,T,hs), bid == b*H + h
#pragma unroll 2
  for (int n = 0; n < 8; ++n) {
    const v8f acc = mm_tile(pf, sVt, n * 16);
#pragma unroll
    for (int j = 0; j < 8; ++j)
      ob[(size_t)(r0 + j + 8 * g) * HS + n * 16 + m] = acc[j];
  }
}

extern "C" void kernel_launch(void* const* d_in, const int* in_sizes, int n_in,
                              void* d_out, int out_size, void* d_ws, size_t ws_size,
                              hipStream_t stream) {
  const float* x  = (const float*)d_in[0];
  const float* Wq = (const float*)d_in[1];
  const float* Wk = (const float*)d_in[2];
  const float* Wv = (const float*)d_in[3];
  float* out = (float*)d_out;
  (void)in_sizes; (void)n_in; (void)out_size; (void)d_ws; (void)ws_size;

  dim3 grid(512 * NHEAD);   // one block per (batch, head)
  dim3 block(256);          // 8 wave32
  hipLaunchKernelGGL(MHFlashAttn_kernel, grid, block, 0, stream,
                     x, Wq, Wk, Wv, out);
}
